// Matcher_79310866088644
// MI455X (gfx1250) — matrix operand order repaired and checked by reading, not compiled, not run on hardware
//
#include <hip/hip_runtime.h>
#include <stdint.h>

// ---------------------------------------------------------------------------
// Problem constants (from the reference)
// ---------------------------------------------------------------------------
#define BATCH 256
#define LLM   128
#define LKG   200
#define EMBD  300
#define EPAD  320   // EMB padded for projection K-loop (10 * 32)
#define HIDN  1024
#define KGP   208   // kg padded for score N-tiles (13 * 16)
#define KGP2  224   // kg padded for attn@v K-loop (7 * 32)

typedef __attribute__((ext_vector_type(16))) __bf16 v16bf;
typedef __attribute__((ext_vector_type(8)))  float  v8f;

union AFrag {
    v16bf          v;
    unsigned short u[16];
    uint4          q[2];
};

union StoreFrag {
    unsigned short s[8];
    uint4          q;
};

__device__ __forceinline__ unsigned short f2bf(float f) {
    unsigned u = __float_as_uint(f);
    unsigned r = 0x7FFFu + ((u >> 16) & 1u);   // round-to-nearest-even
    return (unsigned short)((u + r) >> 16);
}

// ---------------------------------------------------------------------------
// Pack f32 [nBatch*rowsIn, 300] -> bf16 [nBatch*rowsPad, 320]
// (zero-fills the K tail 300..319 and padded rows rowsIn..rowsPad)
// One thread per 2 output elements (one dword store).
// ---------------------------------------------------------------------------
__global__ void __launch_bounds__(256)
pack_bf16(const float* __restrict__ X, unsigned int* __restrict__ out,
          int rowsIn, int rowsPad, long long totalPairs)
{
    const long long p = (long long)blockIdx.x * blockDim.x + threadIdx.x;
    if (p >= totalPairs) return;
    const int       colPair = (int)(p % (EPAD / 2));
    const long long row     = p / (EPAD / 2);
    const int       b       = (int)(row / rowsPad);
    const int       rr      = (int)(row % rowsPad);
    const int       c0      = colPair * 2;
    unsigned lo = 0, hi = 0;
    if (rr < rowsIn) {
        const float* xr = X + ((size_t)b * rowsIn + rr) * EMBD;
        if (c0     < EMBD) lo = f2bf(xr[c0]);
        if (c0 + 1 < EMBD) hi = f2bf(xr[c0 + 1]);
    }
    out[p] = lo | (hi << 16);
}

// ---------------------------------------------------------------------------
// Projection GEMM (all-bf16 inputs, pre-padded):
//   out[m][n] = sum_e Xp[m][e] * Wp[n][e] + bias[n]    (bf16 out)
// Each wave: one 16-row M tile x four 16-col N tiles.  K = 320, 10 WMMA steps
// per N tile, all fragment loads are unconditional 16B vector loads.
// transposed==1 stores out as [b][n][row] with row-dim padded to KGP2.
// ---------------------------------------------------------------------------
__global__ void __launch_bounds__(128)
proj_kernel(const unsigned short* __restrict__ Xp, int rowsPadX,
            const unsigned short* __restrict__ Wp, const float* __restrict__ bias,
            unsigned short* __restrict__ out,
            int rowsPad, int rowsValid, int transposed)
{
    const int lane  = threadIdx.x;               // 0..31
    const int wave  = threadIdx.y;               // 0..3
    const int mTile = blockIdx.x * 4 + wave;
    const int mBase = mTile * 16;
    const int nBase = blockIdx.y * 64;
    const int h     = lane >> 4;
    const int l16   = lane & 15;

    // A-matrix row held by this lane (padded-output row index -> packed-input row)
    const int m  = mBase + l16;
    const int bS = m / rowsPad;
    const int rA = m % rowsPad;
    const uint4* x4 = (const uint4*)(Xp + ((size_t)bS * rowsPadX + rA) * EPAD);

    v8f acc0 = {}, acc1 = {}, acc2 = {}, acc3 = {};
    v8f* acc[4] = { &acc0, &acc1, &acc2, &acc3 };

    for (int k0 = 0; k0 < EPAD; k0 += 32) {
        AFrag a;
        a.q[0] = x4[(k0 + 16 * h) >> 3];
        a.q[1] = x4[(k0 + 16 * h + 8) >> 3];
        #pragma unroll
        for (int i = 0; i < 4; ++i) {
            const int n = nBase + i * 16 + l16;          // B column held by lane
            const uint4* w4 = (const uint4*)(Wp + (size_t)n * EPAD + k0 + 16 * h);
            AFrag bf;
            bf.q[0] = w4[0];
            bf.q[1] = w4[1];
            *acc[i] = __builtin_amdgcn_wmma_f32_16x16x32_bf16(
                          false, a.v, false, bf.v, (short)0, *acc[i], false, false);
        }
    }

    // Epilogue: C/D layout M = r + 8*(lane/16), N = nBase + i*16 + (lane&15).
    const int mg0 = mBase + 8 * h;                 // first of 8 contiguous rows
    const int bb  = mg0 / rowsPad;
    const int rr0 = mg0 % rowsPad;
    #pragma unroll
    for (int i = 0; i < 4; ++i) {
        const int   n  = nBase + i * 16 + l16;
        const float bn = bias[n];
        if (!transposed) {
            #pragma unroll
            for (int r = 0; r < 8; ++r) {
                const float val = ((rr0 + r) < rowsValid) ? ((*acc[i])[r] + bn) : 0.0f;
                out[(size_t)(mg0 + r) * HIDN + n] = f2bf(val);
            }
        } else {
            StoreFrag st;
            #pragma unroll
            for (int r = 0; r < 8; ++r) {
                const float val = ((rr0 + r) < rowsValid) ? ((*acc[i])[r] + bn) : 0.0f;
                st.s[r] = f2bf(val);
            }
            *(uint4*)(out + ((size_t)bb * HIDN + n) * KGP2 + rr0) = st.q;
        }
    }
}

// ---------------------------------------------------------------------------
// Attention kernel: one workgroup per (lm 16-row tile, batch)
//   scores(16x208) -> softmax -> argmax(atomicMax u64) -> aligned(16x1024)
// ---------------------------------------------------------------------------
__global__ void __launch_bounds__(256)
attn_kernel(const unsigned short* __restrict__ qb,
            const unsigned short* __restrict__ kb,
            const unsigned short* __restrict__ vT,
            const int* __restrict__ lm_len,
            float* __restrict__ alignedOut,
            unsigned long long* __restrict__ amax)
{
    __shared__ float          s_sc[16][KGP2];   // scores, then f32 attn
    __shared__ unsigned short s_at[16][KGP2];   // bf16 attn for WMMA
    __shared__ float          s_red[16][16];

    const int tid    = threadIdx.x;
    const int wave   = tid >> 5;
    const int lane   = tid & 31;
    const int h      = lane >> 4;
    const int l16    = lane & 15;
    const int lmTile = blockIdx.x;              // 0..7
    const int b      = blockIdx.y;              // 0..255
    const int len    = lm_len[b];

    // ---- phase 1: scores = q_tile(16x1024) x k^T(1024x208) ----
    const size_t qrow = ((size_t)b * LLM + lmTile * 16 + l16) * HIDN;
    const uint4* q4   = (const uint4*)(qb + qrow);
    for (int t = wave; t < 13; t += 8) {
        const int    nb   = t * 16;
        const size_t krow = ((size_t)b * KGP + nb + l16) * HIDN;
        v8f acc = {};
        for (int k0 = 0; k0 < HIDN; k0 += 32) {
            AFrag a, bf;
            a.q[0] = q4[(k0 + 16 * h) >> 3];
            a.q[1] = q4[(k0 + 16 * h + 8) >> 3];
            const uint4* k4 = (const uint4*)(kb + krow + k0 + 16 * h);
            bf.q[0] = k4[0];
            bf.q[1] = k4[1];
            acc = __builtin_amdgcn_wmma_f32_16x16x32_bf16(
                      false, a.v, false, bf.v, (short)0, acc, false, false);
        }
        #pragma unroll
        for (int r = 0; r < 8; ++r)
            s_sc[8 * h + r][nb + l16] = acc[r];
    }
    __syncthreads();

    // ---- phase 2: softmax over 200 real kg columns (16 threads per row) ----
    {
        const int r = tid >> 4;
        const int c = tid & 15;
        float mx = -3.4e38f;
        for (int j = c; j < LKG; j += 16) mx = fmaxf(mx, s_sc[r][j]);
        s_red[r][c] = mx;
        __syncthreads();
        if (c == 0) {
            float m2 = s_red[r][0];
            #pragma unroll
            for (int i = 1; i < 16; ++i) m2 = fmaxf(m2, s_red[r][i]);
            s_red[r][0] = m2;
        }
        __syncthreads();
        const float rowmax = s_red[r][0];
        float sm = 0.0f;
        for (int j = c; j < LKG; j += 16) sm += __expf(s_sc[r][j] - rowmax);
        __syncthreads();
        s_red[r][c] = sm;
        __syncthreads();
        if (c == 0) {
            float s2 = 0.0f;
            #pragma unroll
            for (int i = 0; i < 16; ++i) s2 += s_red[r][i];
            s_red[r][0] = s2;
        }
        __syncthreads();
        const float inv = 1.0f / s_red[r][0];
        for (int j = c; j < KGP2; j += 16) {
            const float aj = (j < LKG) ? __expf(s_sc[r][j] - rowmax) * inv : 0.0f;
            s_at[r][j] = f2bf(aj);
            s_sc[r][j] = aj;                    // keep f32 attn for argmax
        }
    }
    __syncthreads();

    // ---- phase 3: per-kg-column argmax over valid lm rows of this tile ----
    if (tid < LKG) {
        unsigned long long best = 0ULL;
        #pragma unroll
        for (int r = 0; r < 16; ++r) {
            const int lm = lmTile * 16 + r;
            if (lm >= 1 && lm < len - 1) {
                const unsigned vb = __float_as_uint(s_sc[r][tid]); // attn >= 0
                const unsigned long long key =
                    ((unsigned long long)vb << 32) |
                    (unsigned long long)(0xFFFFFFFFu - (unsigned)lm);
                best = (key > best) ? key : best;
            }
        }
        if (best) atomicMax(&amax[(size_t)b * LKG + tid], best);
    }

    // ---- phase 4: aligned = attn(16x224) x vT^T(224x1024), masked store ----
    const uint4* at4 = (const uint4*)(&s_at[0][0]);
    for (int t = wave; t < 64; t += 8) {
        const int    nb   = t * 16;
        const size_t vrow = ((size_t)b * HIDN + nb + l16) * KGP2;
        v8f acc = {};
        for (int k0 = 0; k0 < KGP2; k0 += 32) {
            AFrag a, bf;
            a.q[0] = at4[(l16 * KGP2 + k0 + 16 * h) >> 3];
            a.q[1] = at4[(l16 * KGP2 + k0 + 16 * h + 8) >> 3];
            const uint4* v4 = (const uint4*)(vT + vrow + k0 + 16 * h);
            bf.q[0] = v4[0];
            bf.q[1] = v4[1];
            acc = __builtin_amdgcn_wmma_f32_16x16x32_bf16(
                      false, a.v, false, bf.v, (short)0, acc, false, false);
        }
        #pragma unroll
        for (int r = 0; r < 8; ++r) {
            const int lm   = lmTile * 16 + 8 * h + r;
            const float mk = (lm >= 1 && lm < len - 1) ? 1.0f : 0.0f;
            alignedOut[((size_t)b * LLM + lm) * HIDN + nb + l16] = acc[r] * mk;
        }
    }
}

// ---------------------------------------------------------------------------
// Glue kernels
// ---------------------------------------------------------------------------
__global__ void amax_init(unsigned long long* amax, int n) {
    const int i = blockIdx.x * blockDim.x + threadIdx.x;
    if (i < n) amax[i] = 0ULL;
}

__global__ void kg_writeback(const unsigned long long* __restrict__ amax,
                             float* __restrict__ out, int n) {
    const int i = blockIdx.x * blockDim.x + threadIdx.x;
    if (i < n) {
        const unsigned long long k = amax[i];
        const unsigned idx = 0xFFFFFFFFu - (unsigned)(k & 0xFFFFFFFFull);
        out[i] = (float)idx;
    }
}

// ---------------------------------------------------------------------------
// Host entry
// ---------------------------------------------------------------------------
extern "C" void kernel_launch(void* const* d_in, const int* in_sizes, int n_in,
                              void* d_out, int out_size, void* d_ws, size_t ws_size,
                              hipStream_t stream)
{
    const float* lm_emb = (const float*)d_in[0];
    const float* kg_emb = (const float*)d_in[1];
    const float* Wq     = (const float*)d_in[2];
    const float* bq     = (const float*)d_in[3];
    const float* Wk     = (const float*)d_in[4];
    const float* bk     = (const float*)d_in[5];
    const float* Wv     = (const float*)d_in[6];
    const float* bv     = (const float*)d_in[7];
    const int*   lm_len = (const int*)d_in[8];

    char* ws = (char*)d_ws;
    const size_t qbBytes  = (size_t)BATCH * LLM  * HIDN * 2;
    const size_t kbBytes  = (size_t)BATCH * KGP  * HIDN * 2;
    const size_t vtBytes  = (size_t)BATCH * HIDN * KGP2 * 2;
    const size_t amBytes  = (size_t)BATCH * LKG * 8;
    const size_t xlmBytes = (size_t)BATCH * LLM  * EPAD * 2;
    const size_t xkgBytes = (size_t)BATCH * KGP2 * EPAD * 2;
    const size_t wBytes   = (size_t)HIDN * EPAD * 2;

    size_t off = 0;
    unsigned short*     qb   = (unsigned short*)(ws + off); off += qbBytes;
    unsigned short*     kb   = (unsigned short*)(ws + off); off += kbBytes;
    unsigned short*     vT   = (unsigned short*)(ws + off); off += vtBytes;
    unsigned long long* amax = (unsigned long long*)(ws + off); off += amBytes;
    unsigned short*     xlm  = (unsigned short*)(ws + off); off += xlmBytes;
    unsigned short*     xkg  = (unsigned short*)(ws + off); off += xkgBytes;
    unsigned short*     wqP  = (unsigned short*)(ws + off); off += wBytes;
    unsigned short*     wkP  = (unsigned short*)(ws + off); off += wBytes;
    unsigned short*     wvP  = (unsigned short*)(ws + off); off += wBytes;

    float* alignedOut = (float*)d_out;
    float* kgOut      = alignedOut + (size_t)BATCH * LLM * HIDN;

    // --- pack inputs to padded bf16 ---
    {
        const long long pLM = (long long)BATCH * LLM * (EPAD / 2);
        pack_bf16<<<dim3((unsigned)((pLM + 255) / 256)), dim3(256), 0, stream>>>(
            lm_emb, (unsigned int*)xlm, LLM, LLM, pLM);
        const long long pKG = (long long)BATCH * KGP2 * (EPAD / 2);
        pack_bf16<<<dim3((unsigned)((pKG + 255) / 256)), dim3(256), 0, stream>>>(
            kg_emb, (unsigned int*)xkg, LKG, KGP2, pKG);
        const long long pW = (long long)HIDN * (EPAD / 2);
        pack_bf16<<<dim3((unsigned)((pW + 255) / 256)), dim3(256), 0, stream>>>(
            Wq, (unsigned int*)wqP, HIDN, HIDN, pW);
        pack_bf16<<<dim3((unsigned)((pW + 255) / 256)), dim3(256), 0, stream>>>(
            Wk, (unsigned int*)wkP, HIDN, HIDN, pW);
        pack_bf16<<<dim3((unsigned)((pW + 255) / 256)), dim3(256), 0, stream>>>(
            Wv, (unsigned int*)wvP, HIDN, HIDN, pW);
    }

    amax_init<<<dim3((BATCH * LKG + 255) / 256), dim3(256), 0, stream>>>(
        amax, BATCH * LKG);

    // Q: -> qb [B*128, 1024]
    proj_kernel<<<dim3((BATCH * LLM) / 64, HIDN / 64), dim3(32, 4), 0, stream>>>(
        xlm, LLM, wqP, bq, qb, LLM, LLM, 0);
    // K: rows padded to 208 -> kb [B*208, 1024]  (input packed at pad 224)
    proj_kernel<<<dim3((BATCH * KGP) / 64, HIDN / 64), dim3(32, 4), 0, stream>>>(
        xkg, KGP2, wkP, bk, kb, KGP, LKG, 0);
    // V: transposed store, kg padded to 224 -> vT [B, 1024, 224]
    proj_kernel<<<dim3((BATCH * KGP2) / 64, HIDN / 64), dim3(32, 4), 0, stream>>>(
        xkg, KGP2, wvP, bv, vT, KGP2, LKG, 1);

    attn_kernel<<<dim3(LLM / 16, BATCH), dim3(256), 0, stream>>>(
        qb, kb, vT, lm_len, alignedOut, amax);

    kg_writeback<<<dim3((BATCH * LKG + 255) / 256), dim3(256), 0, stream>>>(
        amax, kgOut, BATCH * LKG);
}
